// DistanceDecoder_32487132627159
// MI455X (gfx1250) — matrix operand order repaired and checked by reading, not compiled, not run on hardware
//
#include <hip/hip_runtime.h>
#include <hip/hip_bf16.h>

// ---------------------------------------------------------------------------
// DistanceDecoder on MI455X (gfx1250, wave32, WMMA)
//
// Precision: f16 operands / f32 accumulation for all GEMMs; degree norm,
// scatter-add propagation, biases, distance and sigmoid stay in f32.
//
// Blocking: each wave computes a 16x64 C strip (4 WMMA accumulators) so the
// A fragment is loaded once per k-step and reused 4x (8x in the fused dual
// edge-MLP kernel, where the r- and t-MLPs share the same gathered A).
// ---------------------------------------------------------------------------

typedef __attribute__((ext_vector_type(16))) _Float16 v16h;
typedef __attribute__((ext_vector_type(8)))  _Float16 v8h;
typedef __attribute__((ext_vector_type(8)))  float    v8f;

#define LRELU_SLOPE 0.2f

__device__ __forceinline__ v16h cat8h(v8h lo, v8h hi) {
  return __builtin_shufflevector(lo, hi, 0,1,2,3,4,5,6,7,8,9,10,11,12,13,14,15);
}

#define WMMA_F16(a, b, c) \
  __builtin_amdgcn_wmma_f32_16x16x32_f16(false, (a), false, (b), (short)0, (c), false, false)

// ---------------------------- small kernels --------------------------------

__global__ void k_fill_f32(float* __restrict__ p, float v, long n) {
  long i = (long)blockIdx.x * blockDim.x + threadIdx.x;
  if (i < n) p[i] = v;
}

__global__ void k_fill_from_scalar(float* __restrict__ p, const float* __restrict__ v, int n) {
  int i = blockIdx.x * blockDim.x + threadIdx.x;
  if (i < n) p[i] = v[0];
}

__global__ void k_deg_accum(const int* __restrict__ dst, float* __restrict__ deg, int E) {
  int e = blockIdx.x * blockDim.x + threadIdx.x;
  if (e < E) atomicAdd(deg + dst[e], 1.0f);
}

__global__ void k_dinv(const float* __restrict__ deg, float* __restrict__ dinv, int n) {
  int i = blockIdx.x * blockDim.x + threadIdx.x;
  if (i < n) {
    float d = deg[i];
    dinv[i] = (d > 0.0f) ? rsqrtf(d) : 0.0f;
  }
}

__global__ void k_f32_to_f16(const float* __restrict__ x, _Float16* __restrict__ y, long n) {
  long i = (long)blockIdx.x * blockDim.x + threadIdx.x;
  if (i < n) y[i] = (_Float16)x[i];
}

__global__ void k_relu_f32_to_f16(const float* __restrict__ x, _Float16* __restrict__ y, long n) {
  long i = (long)blockIdx.x * blockDim.x + threadIdx.x;
  if (i < n) y[i] = (_Float16)fmaxf(x[i], 0.0f);
}

// W [K,M] row-major f32  ->  Wt [M,K] row-major f16  (B operand, column major)
__global__ void k_transpose_w(const float* __restrict__ W, _Float16* __restrict__ Wt,
                              int K, int M) {
  int i = blockIdx.x * blockDim.x + threadIdx.x;
  if (i < K * M) {
    int k = i / M, m = i % M;
    Wt[(long)m * K + k] = (_Float16)W[i];
  }
}

// Self-loop + bias init:  Out[i,m] = dinv[i]^2 * H[i,m] + b[m]
__global__ void k_out_init(const float* __restrict__ H, const float* __restrict__ dinv,
                           const float* __restrict__ b, float* __restrict__ Out, int M) {
  int i = blockIdx.x, m = threadIdx.x;
  float di = dinv[i];
  Out[(long)i * M + m] = di * di * H[(long)i * M + m] + b[m];
}

// Edge propagation:  Out[dst,m] += dinv[src]*dinv[dst] * H[src,m]
__global__ void k_scatter(const float* __restrict__ H, const int* __restrict__ src,
                          const int* __restrict__ dst, const float* __restrict__ dinv,
                          float* __restrict__ Out, int M) {
  int e = blockIdx.x, m = threadIdx.x;
  int s = src[e], d = dst[e];
  float c = dinv[s] * dinv[d];
  atomicAdd(Out + (long)d * M + m, c * H[(long)s * M + m]);
}

// --------------------------- WMMA GEMM (16x64 strip/wave) -------------------
// C[Mtiles*16, Ncols] = A[Mtiles*16, K] f16 row-major  @  Bt^T
// Bt is [Ncols, K] row-major f16 so each B fragment (lanes 0-15: K=0..15 at
// N=lane; lanes 16-31: K=16..31) is one contiguous 32-byte load per k-step.
// A fragment matches B: lanes 0-15 hold K=0..15, lanes
// 16-31 hold K=16..31, row = lane%16 — loaded ONCE, used by 4 WMMAs.
__global__ void k_wmma_gemm4(const _Float16* __restrict__ A, const _Float16* __restrict__ Bt,
                             float* __restrict__ C, int Mtiles, int Ngroups, int K, int Ncols) {
  int wave = blockIdx.x * (blockDim.x >> 5) + (threadIdx.x >> 5);
  if (wave >= Mtiles * Ngroups) return;                 // wave-uniform
  int tm = wave / Ngroups, tg = wave % Ngroups;
  int lane = threadIdx.x & 31;
  int l16 = lane & 15, hi = lane >> 4;

  const _Float16* ap = A  + (long)(tm * 16 + l16) * K + hi * 16;
  const _Float16* bp = Bt + (long)(tg * 64 + l16) * K + hi * 16;
  const long bstride = (long)16 * K;

  v8f acc[4] = {};
  for (int k0 = 0; k0 < K; k0 += 32) {
    v16h a = cat8h(*(const v8h*)(ap + k0), *(const v8h*)(ap + k0 + 8));
#pragma unroll
    for (int j = 0; j < 4; ++j) {
      v16h b = *(const v16h*)(bp + j * bstride + k0);
      acc[j] = WMMA_F16(a, b, acc[j]);
    }
  }

  float* cp = C + (long)(tm * 16 + hi * 8) * Ncols + tg * 64 + l16;
#pragma unroll
  for (int j = 0; j < 4; ++j)
#pragma unroll
    for (int i = 0; i < 8; ++i)
      cp[(long)i * Ncols + j * 16] = acc[j][i];
}

// ---------------- fused dual edge MLP (r and t share gathered A) ------------
// For each edge e:  feat = [g[src[e]], g[dst[e]]]  (gathered on the fly; each
// 8-half fragment block provably stays on one side of the 128-column seam
// since k0 steps by 32).  Computes BOTH
//   rout[e] += dot(leaky_relu(feat@Wr1+br1), wr2)
//   tout[e] += dot(leaky_relu(feat@Wt1+bt1), wt2)
// rout/tout must be pre-initialized with the second-layer biases.
__global__ void k_edge_mlp2_wmma(const _Float16* __restrict__ g16,
                                 const int* __restrict__ src, const int* __restrict__ dst,
                                 const _Float16* __restrict__ Wr1t, const float* __restrict__ br1,
                                 const float* __restrict__ wr2,
                                 const _Float16* __restrict__ Wt1t, const float* __restrict__ bt1,
                                 const float* __restrict__ wt2,
                                 float* __restrict__ rout, float* __restrict__ tout,
                                 int Etiles) {
  const int K = 256, Ngroups = 4;                       // 256/16 col tiles / 4 per wave
  int wave = blockIdx.x * (blockDim.x >> 5) + (threadIdx.x >> 5);
  if (wave >= Etiles * Ngroups) return;                 // wave-uniform
  int tm = wave / Ngroups, tg = wave % Ngroups;
  int lane = threadIdx.x & 31;
  int l16 = lane & 15, hi = lane >> 4;

  int m = tm * 16 + l16;                                // edge row for this lane's A frag
  long soff = (long)src[m] * 128;
  long doff = (long)dst[m] * 128;
  int kbA = hi * 16;
  const _Float16* brp = Wr1t + (long)(tg * 64 + l16) * K + hi * 16;
  const _Float16* btp = Wt1t + (long)(tg * 64 + l16) * K + hi * 16;
  const long bstride = (long)16 * K;

  v8f accR[4] = {}, accT[4] = {};
  for (int k0 = 0; k0 < K; k0 += 32) {
    int kA1 = k0 + kbA, kA2 = kA1 + 8;
    const _Float16* p1 = g16 + ((kA1 < 128) ? (soff + kA1) : (doff + kA1 - 128));
    const _Float16* p2 = g16 + ((kA2 < 128) ? (soff + kA2) : (doff + kA2 - 128));
    v16h a = cat8h(*(const v8h*)p1, *(const v8h*)p2);
#pragma unroll
    for (int j = 0; j < 4; ++j) {
      v16h b = *(const v16h*)(brp + j * bstride + k0);
      accR[j] = WMMA_F16(a, b, accR[j]);
    }
#pragma unroll
    for (int j = 0; j < 4; ++j) {
      v16h b = *(const v16h*)(btp + j * bstride + k0);
      accT[j] = WMMA_F16(a, b, accT[j]);
    }
  }

  // epilogue: bias + leaky-relu + column-weighted partial sums, then one
  // atomicAdd per row per output
  float pr[8] = {}, pt[8] = {};
  int colbase = tg * 64 + l16;
#pragma unroll
  for (int j = 0; j < 4; ++j) {
    int col = colbase + j * 16;
    float b1r = br1[col], w2r = wr2[col];
    float b1t = bt1[col], w2t = wt2[col];
#pragma unroll
    for (int i = 0; i < 8; ++i) {
      float hr = accR[j][i] + b1r;
      hr = (hr > 0.0f) ? hr : LRELU_SLOPE * hr;
      pr[i] += hr * w2r;
      float ht = accT[j][i] + b1t;
      ht = (ht > 0.0f) ? ht : LRELU_SLOPE * ht;
      pt[i] += ht * w2t;
    }
  }
  int rbase = tm * 16 + hi * 8;
#pragma unroll
  for (int i = 0; i < 8; ++i) {
    atomicAdd(rout + rbase + i, pr[i]);
    atomicAdd(tout + rbase + i, pt[i]);
  }
}

// Final: out[e] = sigmoid( (-||z[s]-z[d]+1e-6|| - r[e]) / t[e] )
__global__ void k_finalize(const float* __restrict__ z, const int* __restrict__ src,
                           const int* __restrict__ dst, const float* __restrict__ r,
                           const float* __restrict__ t, float* __restrict__ out, int E) {
  int e = blockIdx.x * blockDim.x + threadIdx.x;
  if (e >= E) return;
  const float4* zs = (const float4*)(z + (long)src[e] * 128);
  const float4* zd = (const float4*)(z + (long)dst[e] * 128);
  float acc = 0.0f;
#pragma unroll 8
  for (int j = 0; j < 32; ++j) {
    float4 a = zs[j], b = zd[j];
    float d0 = a.x - b.x + 1e-6f;
    float d1 = a.y - b.y + 1e-6f;
    float d2 = a.z - b.z + 1e-6f;
    float d3 = a.w - b.w + 1e-6f;
    acc = fmaf(d0, d0, acc);
    acc = fmaf(d1, d1, acc);
    acc = fmaf(d2, d2, acc);
    acc = fmaf(d3, d3, acc);
  }
  float dist = -sqrtf(acc);
  float x = (dist - r[e]) / t[e];
  out[e] = 1.0f / (1.0f + expf(-x));
}

// ------------------------------- launcher -----------------------------------

static inline size_t align256(size_t x) { return (x + 255) & ~(size_t)255; }

extern "C" void kernel_launch(void* const* d_in, const int* in_sizes, int n_in,
                              void* d_out, int out_size, void* d_ws, size_t ws_size,
                              hipStream_t stream) {
  const int ZD = 128, HD = 256;
  const float* z    = (const float*)d_in[0];
  const int*   eidx = (const int*)d_in[1];
  const float* W0  = (const float*)d_in[2];   const float* b0  = (const float*)d_in[3];
  const float* W1  = (const float*)d_in[4];   const float* b1  = (const float*)d_in[5];
  const float* W2  = (const float*)d_in[6];   const float* b2  = (const float*)d_in[7];
  const float* W3  = (const float*)d_in[8];   const float* b3  = (const float*)d_in[9];
  const float* Wr1 = (const float*)d_in[10];  const float* br1 = (const float*)d_in[11];
  const float* Wr2 = (const float*)d_in[12];  const float* br2 = (const float*)d_in[13];
  const float* Wt1 = (const float*)d_in[14];  const float* bt1 = (const float*)d_in[15];
  const float* Wt2 = (const float*)d_in[16];  const float* bt2 = (const float*)d_in[17];

  const int N = in_sizes[0] / ZD;
  const int E = in_sizes[1] / 2;
  const int* src = eidx;
  const int* dst = eidx + E;

  // ---- workspace carve-out (all 256B aligned) ----
  char* p = (char*)d_ws;
  size_t off = 0;
  float*    deg  = (float*)(p + off);    off = align256(off + (size_t)N * 4);
  float*    dinv = (float*)(p + off);    off = align256(off + (size_t)N * 4);
  float*    H    = (float*)(p + off);    off = align256(off + (size_t)N * HD * 4);
  float*    Out  = (float*)(p + off);    off = align256(off + (size_t)N * HD * 4);
  _Float16* A16  = (_Float16*)(p + off); off = align256(off + (size_t)N * HD * 2);
  _Float16* g16  = (_Float16*)(p + off); off = align256(off + (size_t)N * (HD/2) * 2);
  _Float16* W0t  = (_Float16*)(p + off); off = align256(off + (size_t)HD * ZD * 2);
  _Float16* W1t  = (_Float16*)(p + off); off = align256(off + (size_t)HD * HD * 2);
  _Float16* W2t  = (_Float16*)(p + off); off = align256(off + (size_t)HD * HD * 2);
  _Float16* W3t  = (_Float16*)(p + off); off = align256(off + (size_t)(HD/2) * HD * 2);
  _Float16* Wr1t = (_Float16*)(p + off); off = align256(off + (size_t)HD * HD * 2);
  _Float16* Wt1t = (_Float16*)(p + off); off = align256(off + (size_t)HD * HD * 2);
  float*    rbuf = (float*)(p + off);    off = align256(off + (size_t)E * 4);
  float*    tbuf = (float*)(p + off);    off = align256(off + (size_t)E * 4);
  (void)ws_size; (void)n_in;

  const int T = 256;
  #define CEIL(a,b) (((a)+(b)-1)/(b))

  // ---- degree / normalization ----
  k_fill_f32<<<CEIL((long)N, T), T, 0, stream>>>(deg, 1.0f, N);   // self-loops
  k_deg_accum<<<CEIL(E, T), T, 0, stream>>>(dst, deg, E);
  k_dinv<<<CEIL(N, T), T, 0, stream>>>(deg, dinv, N);

  // ---- weight transpose+convert to f16 [Mout,K] ----
  k_transpose_w<<<CEIL(ZD*HD, T), T, 0, stream>>>(W0,  W0t,  ZD, HD);
  k_transpose_w<<<CEIL(HD*HD, T), T, 0, stream>>>(W1,  W1t,  HD, HD);
  k_transpose_w<<<CEIL(HD*HD, T), T, 0, stream>>>(W2,  W2t,  HD, HD);
  k_transpose_w<<<CEIL(HD*(HD/2), T), T, 0, stream>>>(W3, W3t, HD, HD/2);
  k_transpose_w<<<CEIL(HD*HD, T), T, 0, stream>>>(Wr1, Wr1t, HD, HD);
  k_transpose_w<<<CEIL(HD*HD, T), T, 0, stream>>>(Wt1, Wt1t, HD, HD);

  // ---- input activation ----
  k_f32_to_f16<<<CEIL((long)N*ZD, T), T, 0, stream>>>(z, A16, (long)N * ZD);

  const int Mtiles = N / 16;            // 3125
  const int NG_256 = (HD / 16) / 4;     // 4  (16 col tiles, 4 per wave)
  const int NG_128 = ((HD/2) / 16) / 4; // 2

  // ---- layer 0: [N,128]@[128,256] ----
  k_wmma_gemm4<<<CEIL(Mtiles*NG_256, 8), T, 0, stream>>>(A16, W0t, H, Mtiles, NG_256, ZD, HD);
  k_out_init<<<N, HD, 0, stream>>>(H, dinv, b0, Out, HD);
  k_scatter<<<E, HD, 0, stream>>>(H, src, dst, dinv, Out, HD);
  k_relu_f32_to_f16<<<CEIL((long)N*HD, T), T, 0, stream>>>(Out, A16, (long)N * HD);

  // ---- layer 1: [N,256]@[256,256] ----
  k_wmma_gemm4<<<CEIL(Mtiles*NG_256, 8), T, 0, stream>>>(A16, W1t, H, Mtiles, NG_256, HD, HD);
  k_out_init<<<N, HD, 0, stream>>>(H, dinv, b1, Out, HD);
  k_scatter<<<E, HD, 0, stream>>>(H, src, dst, dinv, Out, HD);
  k_relu_f32_to_f16<<<CEIL((long)N*HD, T), T, 0, stream>>>(Out, A16, (long)N * HD);

  // ---- layer 2: [N,256]@[256,256] ----
  k_wmma_gemm4<<<CEIL(Mtiles*NG_256, 8), T, 0, stream>>>(A16, W2t, H, Mtiles, NG_256, HD, HD);
  k_out_init<<<N, HD, 0, stream>>>(H, dinv, b2, Out, HD);
  k_scatter<<<E, HD, 0, stream>>>(H, src, dst, dinv, Out, HD);
  k_relu_f32_to_f16<<<CEIL((long)N*HD, T), T, 0, stream>>>(Out, A16, (long)N * HD);

  // ---- layer 3: [N,256]@[256,128] -> g ----
  k_wmma_gemm4<<<CEIL(Mtiles*NG_128, 8), T, 0, stream>>>(A16, W3t, H, Mtiles, NG_128, HD, HD/2);
  k_out_init<<<N, HD/2, 0, stream>>>(H, dinv, b3, Out, HD/2);
  k_scatter<<<E, HD/2, 0, stream>>>(H, src, dst, dinv, Out, HD/2);
  k_f32_to_f16<<<CEIL((long)N*(HD/2), T), T, 0, stream>>>(Out, g16, (long)N * (HD/2));

  // ---- fused dual edge MLP (dominant WMMA work, shared gathered A) ----
  const int Etiles = E / 16;            // 50000
  k_fill_from_scalar<<<CEIL(E, T), T, 0, stream>>>(rbuf, br2, E);
  k_fill_from_scalar<<<CEIL(E, T), T, 0, stream>>>(tbuf, bt2, E);
  k_edge_mlp2_wmma<<<CEIL(Etiles*4, 8), T, 0, stream>>>(g16, src, dst,
                                                        Wr1t, br1, Wr2,
                                                        Wt1t, bt1, Wt2,
                                                        rbuf, tbuf, Etiles);

  // ---- distance + sigmoid ----
  k_finalize<<<CEIL(E, T), T, 0, stream>>>(z, src, dst, rbuf, tbuf, (float*)d_out, E);

  #undef CEIL
}
